// WindowAttention3D_7249904796472
// MI455X (gfx1250) — matrix operand order repaired and checked by reading, not compiled, not run on hardware
//
#include <hip/hip_runtime.h>
#include <hip/hip_bf16.h>

// ---------------------------------------------------------------------------
// WindowAttention3D for MI455X (gfx1250), wave32 + v_wmma_f32_16x16x32_f16.
// B=512, N=98, DIM=512, H=16, HD=32.
// Pipeline: f32->f16 cvt, bias gather, QKV GEMM (async-to-LDS double-buffered
// WMMA), per-(b,h) fused attention (QK^T + bias + softmax + PV),
// proj GEMM -> fp32 out.
// ---------------------------------------------------------------------------

typedef __attribute__((ext_vector_type(16))) _Float16 v16h;
typedef __attribute__((ext_vector_type(8)))  float    v8f;

#define NB   512          // batch of windows
#define NT   98           // tokens per window
#define DIM  512
#define NH   16
#define HD   32
#define MTOT (NB * NT)    // 50176 = 392 * 128 exactly

__device__ __forceinline__ v8f wmma16(v16h a, v16h b, v8f c) {
  // D = A(16x32 f16) * B(32x16 f16) + C(16x16 f32)
  return __builtin_amdgcn_wmma_f32_16x16x32_f16(
      /*neg_a=*/false, a, /*neg_b=*/false, b,
      /*c_mod=*/(short)0, c, /*reuse_a=*/false, /*reuse_b=*/false);
}

// A fragment: 16x32 f16 tile, row-major, leading dim ld (halfs).
// Lane m=lane&15 holds row m; K = kb..kb+15 contiguous, kb=16*(lane>=16).
__device__ __forceinline__ v16h frag_a(const _Float16* __restrict__ p, int ld, int lane) {
  const int m  = lane & 15;
  const int kb = (lane & 16) ? 16 : 0;
  const _Float16* r = p + m * ld + kb;
  v16h f;
#pragma unroll
  for (int j = 0; j < 8; ++j) f[j] = r[j];
#pragma unroll
  for (int j = 0; j < 8; ++j) f[8 + j] = r[8 + j];
  return f;
}

// B fragment: 32x16 f16 tile stored n-major (p[n*ld + k]).
// Lane holds column n=lane&15; K = kb..kb+15 contiguous, kb=16*(lane>=16).
__device__ __forceinline__ v16h frag_b(const _Float16* __restrict__ p, int ld, int lane) {
  const int n  = lane & 15;
  const int kb = (lane & 16) ? 16 : 0;
  const _Float16* r = p + n * ld + kb;
  v16h f;
#pragma unroll
  for (int j = 0; j < 16; ++j) f[j] = r[j];
  return f;
}

#define V8F_ZERO v8f{0.f,0.f,0.f,0.f,0.f,0.f,0.f,0.f}

// Async copy 32B (two b128 chunks) global -> LDS; tracked by ASYNCcnt.
// LDS address = low 32 bits of the flat pointer (ISA: LDS aperture truncation).
// INST_OFFSET is added to both the global and the LDS address.
__device__ __forceinline__ void async_copy32(unsigned lds, const _Float16* g) {
  asm volatile(
      "global_load_async_to_lds_b128 %0, %1, off\n\t"
      "global_load_async_to_lds_b128 %0, %1, off offset:16"
      :: "v"(lds), "v"(g) : "memory");
}
__device__ __forceinline__ void wait_async0() {
  asm volatile("s_wait_asynccnt 0x0" ::: "memory");
}

// ---------------------------------------------------------------------------
// Kernel 0: fp32 -> fp16 conversion, vectorized (float4 in, 8B out)
// ---------------------------------------------------------------------------
__global__ void cvt_f32_f16_v4(const float* __restrict__ s, _Float16* __restrict__ d, int n4) {
  int i = blockIdx.x * blockDim.x + threadIdx.x;
  int stride = gridDim.x * blockDim.x;
  for (; i < n4; i += stride) {
    float4 v = ((const float4*)s)[i];
    _Float16 t[4] = {(_Float16)v.x, (_Float16)v.y, (_Float16)v.z, (_Float16)v.w};
    *(uint2*)(d + 4 * (size_t)i) = *(const uint2*)t;
  }
}

// ---------------------------------------------------------------------------
// Kernel 0b: gather relative-position bias once: biasmat[h][n][m]
// ---------------------------------------------------------------------------
__global__ void build_bias(const float* __restrict__ table, const int* __restrict__ idx,
                           float* __restrict__ biasmat) {
  int i = blockIdx.x * blockDim.x + threadIdx.x;  // [0, 16*98*98)
  if (i >= NH * NT * NT) return;
  int h  = i / (NT * NT);
  int nm = i % (NT * NT);
  biasmat[i] = table[idx[nm] * NH + h];
}

// ---------------------------------------------------------------------------
// Shared GEMM mainloop: C[128x64] += A[128x512] * B[512x64], K-step 32.
// A tile: GLOBAL_LOAD_ASYNC_TO_LDS_B128 into ping-pong buffer (no VGPRs).
// B tile: register-staged + transposed to n-major LDS (async can't transpose).
// One s_wait_asynccnt + barrier per k-step; global_prefetch for k+2 tile.
// ---------------------------------------------------------------------------
template <int LDB>
__device__ __forceinline__ void gemm_mainloop(
    const _Float16* __restrict__ A, const _Float16* __restrict__ Bsrc,
    int tm, int tn, int tid, int w, int lane,
    _Float16 (*As)[128][40], _Float16 (*Bs)[64][40], v8f acc[4])
{
  const int arow = tid >> 1, ach = (tid & 1) * 16;
  const int bk = tid >> 3, bng = (tid & 7) * 8;
  const _Float16* aptr = A + (size_t)(tm * 128 + arow) * DIM + ach;
  const _Float16* bptr = Bsrc + (size_t)bk * LDB + tn * 64 + bng;
  const unsigned ldsA0 = (unsigned)(size_t)(void*)&As[0][arow][ach];
  const unsigned ldsAStride = (unsigned)((char*)&As[1][0][0] - (char*)&As[0][0][0]);

  // preload tile 0: A async -> LDS[0]; B via registers, transposed
  {
    async_copy32(ldsA0, aptr);
    uint4 b0 = *(const uint4*)(bptr);
    _Float16 t[8]; *(uint4*)t = b0;
#pragma unroll
    for (int i = 0; i < 8; ++i) Bs[0][bng + i][bk] = t[i];
    wait_async0();
  }
  __syncthreads();

  int buf = 0;
  for (int kt = 0; kt < 16; ++kt) {
    const bool nxt = (kt + 1) < 16;
    uint4 b0;
    if (nxt) {
      // A tile k+1: async straight into the other LDS buffer
      async_copy32(ldsA0 + (buf ^ 1) * ldsAStride, aptr + (kt + 1) * 32);
      // B tile k+1: stage in registers (transposed store below)
      b0 = *(const uint4*)(bptr + (size_t)(kt + 1) * 32 * LDB);
      if (kt + 2 < 16) {
        __builtin_prefetch(aptr + (kt + 2) * 32, 0, 1);           // global_prefetch_b8
        __builtin_prefetch(bptr + (size_t)(kt + 2) * 32 * LDB, 0, 1);
      }
    }

    v16h a = frag_a(&As[buf][w * 16][0], 40, lane);
#pragma unroll
    for (int nt = 0; nt < 4; ++nt) {
      v16h b = frag_b(&Bs[buf][nt * 16][0], 40, lane);
      acc[nt] = wmma16(a, b, acc[nt]);
    }

    if (nxt) {
      const int nb = buf ^ 1;
      _Float16 t[8]; *(uint4*)t = b0;
#pragma unroll
      for (int i = 0; i < 8; ++i) Bs[nb][bng + i][bk] = t[i];
      wait_async0();       // own async A copies landed in LDS
      __syncthreads();     // block-uniform condition -> legal
    }
    buf ^= 1;
  }
}

// ---------------------------------------------------------------------------
// Kernel 1: QKV GEMM.  xh[50176,512]f16 @ wh[512,1536]f16 + qkv_b
// Epilogue scatters to q/k/v [B,H,N,HD] f16, q pre-scaled by HD^-0.5.
// ---------------------------------------------------------------------------
__global__ __launch_bounds__(256) void qkv_gemm(
    const _Float16* __restrict__ xh, const _Float16* __restrict__ wh,
    const float* __restrict__ qkvb,
    _Float16* __restrict__ qh, _Float16* __restrict__ kh, _Float16* __restrict__ vh)
{
  __shared__ __align__(16) _Float16 As[2][128][40];
  __shared__ __align__(16) _Float16 Bs[2][64][40];
  const int tn = blockIdx.x, tm = blockIdx.y;
  const int tid = threadIdx.x, w = tid >> 5, lane = tid & 31;

  v8f acc[4];
#pragma unroll
  for (int i = 0; i < 4; ++i) acc[i] = V8F_ZERO;

  gemm_mainloop<3 * DIM>(xh, wh, tm, tn, tid, w, lane, As, Bs, acc);

  const float scale = 0.17677669529663687f;  // HD^-0.5
  const int n = lane & 15, rbase = (lane & 16) ? 8 : 0;

  // one division for 8 consecutive rows, then increment-with-wrap
  int bbs[8], toks[8];
  {
    int mg0 = tm * 128 + w * 16 + rbase;
    int bb = mg0 / NT, tok = mg0 - bb * NT;
#pragma unroll
    for (int r = 0; r < 8; ++r) {
      bbs[r] = bb; toks[r] = tok;
      if (++tok == NT) { tok = 0; ++bb; }
    }
  }

#pragma unroll
  for (int nt = 0; nt < 4; ++nt) {
    int ng  = tn * 64 + nt * 16 + n;       // [0,1536)
    int t   = ng >> 9;                     // 0=q 1=k 2=v
    int rem = ng & 511;
    int hh  = rem >> 5, hd = rem & 31;
    _Float16* dst = (t == 0) ? qh : ((t == 1) ? kh : vh);
    float bias = qkvb[ng];
    float sc   = (t == 0) ? scale : 1.f;
#pragma unroll
    for (int r = 0; r < 8; ++r) {
      float v = (acc[nt][r] + bias) * sc;
      dst[(((size_t)(bbs[r] * NH + hh) * NT + toks[r]) << 5) + hd] = (_Float16)v;
    }
  }
}

// ---------------------------------------------------------------------------
// Kernel 2: attention for one (b,h): S = q k^T (+bias), softmax, out = P v.
// 98 tokens padded to 112 (7 tiles of 16); K of P@V padded to 128 (4 steps).
// ---------------------------------------------------------------------------
__global__ __launch_bounds__(256) void attn_kernel(
    const _Float16* __restrict__ qh, const _Float16* __restrict__ kh,
    const _Float16* __restrict__ vh, const float* __restrict__ biasmat,
    _Float16* __restrict__ xo)
{
  __shared__ float S[112][132];   // scores, then probabilities (f32)
  const int bh = blockIdx.x;
  const int b = bh >> 4, h = bh & 15;
  const _Float16* qb = qh + (size_t)bh * NT * HD;
  const _Float16* kb = kh + (size_t)bh * NT * HD;
  const _Float16* vb = vh + (size_t)bh * NT * HD;
  const int tid = threadIdx.x, w = tid >> 5, lane = tid & 31;

  // ---- S = q k^T : waves 0..6 own one 16-row tile each, 7 column tiles ----
  if (w < 7) {
    v16h a = frag_a(qb + w * 16 * HD, HD, lane);   // HD=32 -> one K step
    const int n = lane & 15, rbase = (lane & 16) ? 8 : 0;
#pragma unroll
    for (int ct = 0; ct < 7; ++ct) {
      // k row-major [tok][HD] is exactly the n-major B layout
      v16h bf = frag_b(kb + ct * 16 * HD, HD, lane);
      v8f c = V8F_ZERO;
      c = wmma16(a, bf, c);
#pragma unroll
      for (int r = 0; r < 8; ++r) S[w * 16 + rbase + r][ct * 16 + n] = c[r];
    }
  }
  __syncthreads();

  // ---- bias + softmax (one thread per row; pad rows/cols zeroed) ----
  if (tid < 112) {
    const int row = tid;
    if (row < NT) {
      const float* brow = biasmat + (size_t)h * (NT * NT) + row * NT;
      float mx = -3.4e38f;
      for (int m = 0; m < NT; ++m) {
        float v = S[row][m] + brow[m];
        S[row][m] = v;
        mx = fmaxf(mx, v);
      }
      float sum = 0.f;
      for (int m = 0; m < NT; ++m) {
        float e = __expf(S[row][m] - mx);
        S[row][m] = e;
        sum += e;
      }
      float inv = 1.f / sum;
      for (int m = 0; m < NT; ++m) S[row][m] *= inv;
      for (int m = NT; m < 128; ++m) S[row][m] = 0.f;
    } else {
      for (int m = 0; m < 128; ++m) S[row][m] = 0.f;
    }
  }
  __syncthreads();

  // ---- out = P @ v : 7 row tiles x 2 col tiles (HD=32), K padded to 128 ----
  if (w < 7) {
    v8f acc0 = V8F_ZERO, acc1 = V8F_ZERO;
    const int n = lane & 15;
    const int ka  = (lane & 16) ? 16 : 0;  // A K-offset pattern
    const int kbb = (lane & 16) ? 16 : 0;  // B K-offset pattern
    const int m = lane & 15;
#pragma unroll
    for (int ks = 0; ks < 4; ++ks) {
      v16h a;
      const float* pr = &S[w * 16 + m][ks * 32 + ka];
#pragma unroll
      for (int j = 0; j < 8; ++j) {
        a[j]     = (_Float16)pr[j];
        a[8 + j] = (_Float16)pr[8 + j];
      }
      v16h b0, b1;
#pragma unroll
      for (int j = 0; j < 16; ++j) {
        int tok = ks * 32 + kbb + j;
        bool ok = tok < NT;                        // mask pad rows (avoid NaN*0)
        b0[j] = ok ? vb[tok * HD + n]      : (_Float16)0.f;
        b1[j] = ok ? vb[tok * HD + 16 + n] : (_Float16)0.f;
      }
      acc0 = wmma16(a, b0, acc0);
      acc1 = wmma16(a, b1, acc1);
    }
    const int rbase = (lane & 16) ? 8 : 0;
#pragma unroll
    for (int r = 0; r < 8; ++r) {
      int tok = w * 16 + rbase + r;
      if (tok < NT) {
        _Float16* o = xo + ((size_t)(b * NT + tok) << 9) + h * HD;  // [B,N,DIM]
        o[n]      = (_Float16)acc0[r];
        o[16 + n] = (_Float16)acc1[r];
      }
    }
  }
}

// ---------------------------------------------------------------------------
// Kernel 3: output projection.  xo[50176,512]f16 @ projw[512,512]f16 + proj_b
// ---------------------------------------------------------------------------
__global__ __launch_bounds__(256) void proj_gemm(
    const _Float16* __restrict__ xo, const _Float16* __restrict__ wh,
    const float* __restrict__ pb, float* __restrict__ out)
{
  __shared__ __align__(16) _Float16 As[2][128][40];
  __shared__ __align__(16) _Float16 Bs[2][64][40];
  const int tn = blockIdx.x, tm = blockIdx.y;
  const int tid = threadIdx.x, w = tid >> 5, lane = tid & 31;

  v8f acc[4];
#pragma unroll
  for (int i = 0; i < 4; ++i) acc[i] = V8F_ZERO;

  gemm_mainloop<DIM>(xo, wh, tm, tn, tid, w, lane, As, Bs, acc);

  const int n = lane & 15, rbase = (lane & 16) ? 8 : 0;
#pragma unroll
  for (int nt = 0; nt < 4; ++nt) {
    int ng = tn * 64 + nt * 16 + n;
    float bias = pb[ng];
#pragma unroll
    for (int r = 0; r < 8; ++r) {
      int mg = tm * 128 + w * 16 + rbase + r;
      out[(size_t)mg * DIM + ng] = acc[nt][r] + bias;
    }
  }
}

// ---------------------------------------------------------------------------
// Host launcher
// ---------------------------------------------------------------------------
extern "C" void kernel_launch(void* const* d_in, const int* in_sizes, int n_in,
                              void* d_out, int out_size, void* d_ws, size_t ws_size,
                              hipStream_t stream) {
  const float* x          = (const float*)d_in[0];   // [512,98,512]
  const float* qkv_w      = (const float*)d_in[1];   // [512,1536]
  const float* qkv_b      = (const float*)d_in[2];   // [1536]
  const float* proj_w     = (const float*)d_in[3];   // [512,512]
  const float* proj_b     = (const float*)d_in[4];   // [512]
  const float* bias_table = (const float*)d_in[5];   // [507,16]
  const int*   rel_index  = (const int*)d_in[6];     // [98,98]
  float* out = (float*)d_out;

  char* ws = (char*)d_ws;
  size_t off = 0;
  _Float16* qkvw_h = (_Float16*)(ws + off); off += (size_t)DIM * 3 * DIM * 2;   // 1.5 MB
  _Float16* projw_h= (_Float16*)(ws + off); off += (size_t)DIM * DIM * 2;       // 0.5 MB
  _Float16* xh     = (_Float16*)(ws + off); off += (size_t)MTOT * DIM * 2;      // 51 MB
  _Float16* qh     = (_Float16*)(ws + off); off += (size_t)NB * NH * NT * HD * 2;
  _Float16* kh     = (_Float16*)(ws + off); off += (size_t)NB * NH * NT * HD * 2;
  _Float16* vh     = (_Float16*)(ws + off); off += (size_t)NB * NH * NT * HD * 2;
  float*    biasm  = (float*)(ws + off);    off += (size_t)NH * NT * NT * 4;    // 0.6 MB
  _Float16* xo     = (_Float16*)(ws + off); off += (size_t)MTOT * DIM * 2;      // 51 MB
  // total ws usage ~248 MB (q/k/v padded-row overreads stay inside ws)

  // 0) conversions + bias matrix
  cvt_f32_f16_v4<<<2048, 256, 0, stream>>>(x, xh, (MTOT * DIM) / 4);
  cvt_f32_f16_v4<<<256, 256, 0, stream>>>(qkv_w, qkvw_h, (DIM * 3 * DIM) / 4);
  cvt_f32_f16_v4<<<256, 256, 0, stream>>>(proj_w, projw_h, (DIM * DIM) / 4);
  build_bias<<<(NH * NT * NT + 255) / 256, 256, 0, stream>>>(bias_table, rel_index, biasm);

  // 1) QKV projection: grid = (1536/64, 50176/128)
  qkv_gemm<<<dim3(24, 392), 256, 0, stream>>>(xh, qkvw_h, qkv_b, qh, kh, vh);

  // 2) attention: one block per (b,h)
  attn_kernel<<<NB * NH, 256, 0, stream>>>(qh, kh, vh, biasm, xo);

  // 3) output projection: grid = (512/64, 50176/128)
  proj_gemm<<<dim3(8, 392), 256, 0, stream>>>(xo, projw_h, proj_b, out);
}
